// EncoderLayer_88476326298146
// MI455X (gfx1250) — compile-verified, matrix operand order checked by reading, not run on hardware
//
#include <hip/hip_runtime.h>
#include <math.h>

// Problem constants (fixed by the reference).
#define L_N   16384
#define E_N   131072
#define C_N   512
#define H_N   8
#define DH_N  64
#define HID_N 1024

typedef __attribute__((ext_vector_type(16))) __bf16 v16bf;
typedef __attribute__((ext_vector_type(8)))  __bf16 v8bf;
typedef __attribute__((ext_vector_type(8)))  float  v8f;

// ---------------------------------------------------------------------------
// Weight packing: fp32 W[K x N] row-major -> bf16 WMMA B-fragments.
// Fragment f = (kt, nt) covers K rows [32*kt, +32), N cols [16*nt, +16).
// Within a fragment: 512 bf16, lane-major (32 lanes x 16 elems).
//   lane l: col n = 16*nt + (l&15); hg = l>>4
//   elem j: k = 32*kt + (j&7) + 8*hg + 16*(j>>3)
// (mirrors the 16-bit A-matrix layout in CDNA5 ISA sec 7.12.2)
// ---------------------------------------------------------------------------
__global__ __launch_bounds__(256) void k_pack_w(const float* __restrict__ W,
                                                __bf16* __restrict__ Bp,
                                                int K, int N) {
  size_t i = (size_t)blockIdx.x * 256 + threadIdx.x;
  if (i >= (size_t)K * N) return;
  int   e  = (int)(i & 511);
  size_t f = i >> 9;
  int nfr  = N >> 4;
  int nt   = (int)(f % nfr);
  int kt   = (int)(f / nfr);
  int l    = e >> 4;
  int j    = e & 15;
  int hg   = l >> 4;
  int n    = (nt << 4) + (l & 15);
  int k    = (kt << 5) + (j & 7) + 8 * hg + ((j >> 3) << 4);
  Bp[i] = (__bf16)W[(size_t)k * N + n];
}

// ---------------------------------------------------------------------------
// LayerNorm over C=512, output bf16. One 256-thread block per row.
// ---------------------------------------------------------------------------
__global__ __launch_bounds__(256) void k_ln_bf16(const float* __restrict__ x,
                                                 const float* __restrict__ g,
                                                 const float* __restrict__ b,
                                                 __bf16* __restrict__ z) {
  const int t = threadIdx.x;
  const float* xr = x + (size_t)blockIdx.x * C_N;
  __shared__ float sm[256];
  float a0 = xr[t], a1 = xr[t + 256];
  sm[t] = a0 + a1;
  __syncthreads();
  for (int o = 128; o > 0; o >>= 1) {
    if (t < o) sm[t] += sm[t + o];
    __syncthreads();
  }
  float mu = sm[0] * (1.0f / C_N);
  __syncthreads();
  float d0 = a0 - mu, d1 = a1 - mu;
  sm[t] = d0 * d0 + d1 * d1;
  __syncthreads();
  for (int o = 128; o > 0; o >>= 1) {
    if (t < o) sm[t] += sm[t + o];
    __syncthreads();
  }
  float rs = rsqrtf(sm[0] * (1.0f / C_N) + 1e-5f);
  __bf16* zr = z + (size_t)blockIdx.x * C_N;
  zr[t]       = (__bf16)(d0 * rs * g[t] + b[t]);
  zr[t + 256] = (__bf16)(d1 * rs * g[t + 256] + b[t + 256]);
}

// ---------------------------------------------------------------------------
// bf16 WMMA GEMM: out[M x N] = A[M x K] @ W[K x N] + bias (+res / gelu).
// Block = 256 threads = 8 wave32 waves; wave w owns a 32x64 tile:
//   rows [32*by, +32)  (2 M-subtiles), cols [(bx*8+w)*64, +64) (4 N-subtiles).
// All fragment loads of one K-step go to distinct registers so the compiler
// can clause the 12 global_load_b128 and overlap them with the 8 WMMAs.
// MODE 0: fp32 out = acc + bias
// MODE 1: fp32 out = acc + bias + res
// MODE 2: bf16 out = gelu(acc + bias)   (tanh approximation, as jax.nn.gelu)
// ---------------------------------------------------------------------------
__device__ __forceinline__ v16bf ld_frag(const __bf16* p, int elem_stride16) {
  v8bf lo = *(const v8bf*)(p);
  v8bf hi = *(const v8bf*)(p + elem_stride16);
  return __builtin_shufflevector(lo, hi, 0, 1, 2, 3, 4, 5, 6, 7,
                                 8, 9, 10, 11, 12, 13, 14, 15);
}

template <int MODE>
__global__ __launch_bounds__(256) void k_gemm_bf16(
    const __bf16* __restrict__ A, const __bf16* __restrict__ Bp,
    const float* __restrict__ bias, const float* __restrict__ res,
    float* __restrict__ outf, __bf16* __restrict__ outh,
    int M, int N, int K) {
  const int lane  = threadIdx.x & 31;
  const int wave  = threadIdx.x >> 5;
  const int hg    = lane >> 4;
  const int l16   = lane & 15;
  const int tileM = blockIdx.y << 5;                 // 32 rows per block
  const int tileN = (blockIdx.x * 8 + wave) << 6;    // 64 cols per wave
  if (tileN >= N) return;  // wave-uniform

  v8f acc[2][4];
#pragma unroll
  for (int mi = 0; mi < 2; ++mi)
#pragma unroll
    for (int s = 0; s < 4; ++s) acc[mi][s] = (v8f){};

  const __bf16* Arow0 = A + (size_t)(tileM + l16) * K + 8 * hg;
  const __bf16* Arow1 = Arow0 + (size_t)16 * K;
  const int nfrag = N >> 4;

  for (int kb = 0; kb < K; kb += 32) {
    // A fragments: lane<16 -> K {0..7,16..23}; lane>=16 -> K {8..15,24..31}
    v16bf a0 = ld_frag(Arow0 + kb, 16);
    v16bf a1 = ld_frag(Arow1 + kb, 16);
    const __bf16* bb =
        Bp + (((size_t)(kb >> 5) * nfrag + (tileN >> 4)) << 9) + lane * 16;
    v16bf b0 = ld_frag(bb, 8);
    v16bf b1 = ld_frag(bb + 512, 8);
    v16bf b2 = ld_frag(bb + 1024, 8);
    v16bf b3 = ld_frag(bb + 1536, 8);
    if (kb + 32 < K) {
      __builtin_prefetch(Arow0 + kb + 32, 0, 3);
      __builtin_prefetch(Arow1 + kb + 32, 0, 3);
      __builtin_prefetch(bb + ((size_t)nfrag << 9), 0, 3);
    }
    acc[0][0] = __builtin_amdgcn_wmma_f32_16x16x32_bf16(false, a0, false, b0, (short)0, acc[0][0], false, false);
    acc[0][1] = __builtin_amdgcn_wmma_f32_16x16x32_bf16(false, a0, false, b1, (short)0, acc[0][1], false, false);
    acc[0][2] = __builtin_amdgcn_wmma_f32_16x16x32_bf16(false, a0, false, b2, (short)0, acc[0][2], false, false);
    acc[0][3] = __builtin_amdgcn_wmma_f32_16x16x32_bf16(false, a0, false, b3, (short)0, acc[0][3], false, false);
    acc[1][0] = __builtin_amdgcn_wmma_f32_16x16x32_bf16(false, a1, false, b0, (short)0, acc[1][0], false, false);
    acc[1][1] = __builtin_amdgcn_wmma_f32_16x16x32_bf16(false, a1, false, b1, (short)0, acc[1][1], false, false);
    acc[1][2] = __builtin_amdgcn_wmma_f32_16x16x32_bf16(false, a1, false, b2, (short)0, acc[1][2], false, false);
    acc[1][3] = __builtin_amdgcn_wmma_f32_16x16x32_bf16(false, a1, false, b3, (short)0, acc[1][3], false, false);
  }

  // Epilogue.  C layout: lane<16 -> M = r, lane>=16 -> M = 8+r; col = lane&15.
#pragma unroll
  for (int mi = 0; mi < 2; ++mi) {
#pragma unroll
    for (int s = 0; s < 4; ++s) {
      int col = tileN + (s << 4) + l16;
      float bv = bias[col];
#pragma unroll
      for (int r = 0; r < 8; ++r) {
        int row = tileM + 16 * mi + r + 8 * hg;
        size_t idx = (size_t)row * N + col;
        float v = acc[mi][s][r] + bv;
        if (MODE == 1) v += res[idx];
        if (MODE == 2) {
          float u = 0.7978845608028654f * (v + 0.044715f * v * v * v);
          outh[idx] = (__bf16)(0.5f * v * (1.0f + tanhf(u)));
        } else {
          outf[idx] = v;
        }
      }
    }
  }
}

// ---------------------------------------------------------------------------
// Edge attention passes.
// ---------------------------------------------------------------------------
__device__ __forceinline__ void atomicMaxF(float* addr, float v) {
  if (v >= 0.0f) atomicMax((int*)addr, __float_as_int(v));
  else           atomicMin((unsigned int*)addr, __float_as_uint(v));
}

__global__ __launch_bounds__(256) void k_init_mden(float* __restrict__ m,
                                                   float* __restrict__ den) {
  int i = blockIdx.x * 256 + threadIdx.x;
  if (i < L_N * H_N) { m[i] = -__builtin_inff(); den[i] = 0.0f; }
}

__global__ __launch_bounds__(256) void k_zero_f32(float* __restrict__ p, int n) {
  int i = blockIdx.x * 256 + threadIdx.x;
  if (i < n) p[i] = 0.0f;
}

// Pass A: score[e,h] = -||q_e - k_e||^2 / sqrt(DH) + pos_att_bias; row max.
__global__ __launch_bounds__(256) void k_edge_score(
    const float* __restrict__ q, const float* __restrict__ k,
    const int* __restrict__ row_index, const int* __restrict__ col_index,
    const float* __restrict__ pos_bias, float* __restrict__ score,
    float* __restrict__ m) {
  int t = blockIdx.x * 256 + threadIdx.x;
  if (t >= E_N * H_N) return;
  int e = t >> 3, h = t & 7;
  int r = row_index[e], c = col_index[e];
  const float4* qp = (const float4*)(q + (size_t)r * C_N + h * DH_N);
  const float4* kp = (const float4*)(k + (size_t)c * C_N + h * DH_N);
  float s = 0.0f;
#pragma unroll
  for (int d = 0; d < DH_N / 4; ++d) {
    float4 qa = qp[d], ka = kp[d];
    float dx = qa.x - ka.x, dy = qa.y - ka.y, dz = qa.z - ka.z, dw = qa.w - ka.w;
    s += dx * dx + dy * dy + dz * dz + dw * dw;
  }
  s = -s * 0.125f + pos_bias[t];  // 1/sqrt(64) = 0.125
  score[t] = s;
  atomicMaxF(&m[r * H_N + h], s);
}

// Pass B: pexp = exp(score - m[row]); den += pexp  (in-place over score buf).
__global__ __launch_bounds__(256) void k_edge_pexp(
    const int* __restrict__ row_index, const float* __restrict__ m,
    float* __restrict__ score, float* __restrict__ den) {
  int t = blockIdx.x * 256 + threadIdx.x;
  if (t >= E_N * H_N) return;
  int e = t >> 3, h = t & 7;
  int r = row_index[e];
  float mm = m[r * H_N + h];
  if (!(mm > -3.0e38f)) mm = 0.0f;  // isfinite(m) ? m : 0
  float p = __expf(score[t] - mm);
  score[t] = p;
  atomicAdd(&den[r * H_N + h], p);
}

// Pass C: attout[row, c] += alpha * (v[col, c] + rel . Wvec[:, c] + bvec[c]).
// One thread per (edge, 4-channel group): float4 loads, 4 f32 atomic adds.
__global__ __launch_bounds__(256) void k_edge_accum(
    const float* __restrict__ v, const int* __restrict__ row_index,
    const int* __restrict__ col_index, const int* __restrict__ to_col_index,
    const float* __restrict__ dist, const float* __restrict__ pos,
    const float* __restrict__ col_pos, const float* __restrict__ Wvec,
    const float* __restrict__ bvec, const float* __restrict__ pexp,
    const float* __restrict__ den, float* __restrict__ attout) {
  size_t t = (size_t)blockIdx.x * 256 + threadIdx.x;
  if (t >= (size_t)E_N * (C_N / 4)) return;
  int e = (int)(t >> 7);            // 128 channel-groups per edge
  int c = (int)(t & 127) << 2;      // channel base (multiple of 4)
  int h = c >> 6;
  int r  = row_index[e];
  int ci = col_index[e];
  int tc = to_col_index[e];
  float invd = 1.0f / dist[e];
  float rx = (col_pos[tc * 3 + 0] - pos[r * 3 + 0]) * invd;
  float ry = (col_pos[tc * 3 + 1] - pos[r * 3 + 1]) * invd;
  float rz = (col_pos[tc * 3 + 2] - pos[r * 3 + 2]) * invd;
  float dn = den[r * H_N + h];
  if (dn == 0.0f) dn = 1.0f;
  float alpha = pexp[e * H_N + h] / dn;
  float4 vv = *(const float4*)(v + (size_t)ci * C_N + c);
  float4 w0 = *(const float4*)(Wvec + c);
  float4 w1 = *(const float4*)(Wvec + C_N + c);
  float4 w2 = *(const float4*)(Wvec + 2 * C_N + c);
  float4 bb = *(const float4*)(bvec + c);
  float o0 = alpha * (vv.x + rx * w0.x + ry * w1.x + rz * w2.x + bb.x);
  float o1 = alpha * (vv.y + rx * w0.y + ry * w1.y + rz * w2.y + bb.y);
  float o2 = alpha * (vv.z + rx * w0.z + ry * w1.z + rz * w2.z + bb.z);
  float o3 = alpha * (vv.w + rx * w0.w + ry * w1.w + rz * w2.w + bb.w);
  float* ap = attout + (size_t)r * C_N + c;
  atomicAdd(ap + 0, o0);
  atomicAdd(ap + 1, o1);
  atomicAdd(ap + 2, o2);
  atomicAdd(ap + 3, o3);
}

__global__ __launch_bounds__(256) void k_f32_to_bf16(const float* __restrict__ s,
                                                     __bf16* __restrict__ d,
                                                     int n) {
  int i = blockIdx.x * 256 + threadIdx.x;
  if (i < n) d[i] = (__bf16)s[i];
}

// ---------------------------------------------------------------------------
// Host-side orchestration.
// ---------------------------------------------------------------------------
extern "C" void kernel_launch(void* const* d_in, const int* in_sizes, int n_in,
                              void* d_out, int out_size, void* d_ws, size_t ws_size,
                              hipStream_t stream) {
  const float* x        = (const float*)d_in[0];
  const int*   row_idx  = (const int*)d_in[1];
  const int*   col_idx  = (const int*)d_in[2];
  const int*   tcol_idx = (const int*)d_in[3];
  // d_in[4] (att_bias) is unused by the reference.
  const float* pos_bias = (const float*)d_in[5];
  const float* dist     = (const float*)d_in[6];
  const float* pos      = (const float*)d_in[7];
  const float* col_pos  = (const float*)d_in[8];
  const float* ln1_g    = (const float*)d_in[9];
  const float* ln1_b    = (const float*)d_in[10];
  const float* ln2_g    = (const float*)d_in[11];
  const float* ln2_b    = (const float*)d_in[12];
  const float* Wq = (const float*)d_in[13]; const float* bq = (const float*)d_in[14];
  const float* Wk = (const float*)d_in[15]; const float* bk = (const float*)d_in[16];
  const float* Wv = (const float*)d_in[17]; const float* bv = (const float*)d_in[18];
  const float* Wvec = (const float*)d_in[19]; const float* bvec = (const float*)d_in[20];
  const float* Wo = (const float*)d_in[21]; const float* bo = (const float*)d_in[22];
  const float* W1 = (const float*)d_in[23]; const float* b1 = (const float*)d_in[24];
  const float* W2 = (const float*)d_in[25]; const float* b2 = (const float*)d_in[26];
  float* out = (float*)d_out;

  // Workspace carve-up (aliased across phases; all offsets 256B-aligned).
  char* w = (char*)d_ws;
  __bf16* zb   = (__bf16*)w; w += (size_t)L_N * C_N * 2;  // z -> attout_bf16 -> z2
  float*  bufA = (float*)w;  w += (size_t)L_N * C_N * 4;  // q -> attout
  float*  bufB = (float*)w;  w += (size_t)L_N * C_N * 4;  // k -> x1
  float*  bufC = (float*)w;  w += (size_t)L_N * C_N * 4;  // v -> ffn hidden (bf16)
  float*  score = (float*)w; w += (size_t)E_N * H_N * 4;  // score -> pexp
  float*  mrow  = (float*)w; w += (size_t)L_N * H_N * 4;
  float*  den   = (float*)w; w += (size_t)L_N * H_N * 4;
  __bf16* WqP = (__bf16*)w; w += (size_t)C_N * C_N * 2;
  __bf16* WkP = (__bf16*)w; w += (size_t)C_N * C_N * 2;
  __bf16* WvP = (__bf16*)w; w += (size_t)C_N * C_N * 2;
  __bf16* WoP = (__bf16*)w; w += (size_t)C_N * C_N * 2;
  __bf16* W1P = (__bf16*)w; w += (size_t)C_N * HID_N * 2;
  __bf16* W2P = (__bf16*)w; w += (size_t)HID_N * C_N * 2;
  __bf16* hb  = (__bf16*)bufC;  // ffn hidden aliases v (dead after edge accum)

  const int nCC = C_N * C_N, nCH = C_N * HID_N;
  dim3 b256(256);

  // 0) init softmax stats
  k_init_mden<<<(L_N * H_N + 255) / 256, b256, 0, stream>>>(mrow, den);

  // 1) pack weights into WMMA fragment layout
  k_pack_w<<<(nCC + 255) / 256, b256, 0, stream>>>(Wq, WqP, C_N, C_N);
  k_pack_w<<<(nCC + 255) / 256, b256, 0, stream>>>(Wk, WkP, C_N, C_N);
  k_pack_w<<<(nCC + 255) / 256, b256, 0, stream>>>(Wv, WvP, C_N, C_N);
  k_pack_w<<<(nCC + 255) / 256, b256, 0, stream>>>(Wo, WoP, C_N, C_N);
  k_pack_w<<<(nCH + 255) / 256, b256, 0, stream>>>(W1, W1P, C_N, HID_N);
  k_pack_w<<<(nCH + 255) / 256, b256, 0, stream>>>(W2, W2P, HID_N, C_N);

  // 2) z = LN1(x)  (bf16)
  k_ln_bf16<<<L_N, b256, 0, stream>>>(x, ln1_g, ln1_b, zb);

  // 3) q,k,v GEMMs (WMMA bf16 -> fp32); 32-row block tiles
  dim3 g512(1, L_N / 32), g1024(2, L_N / 32);
  k_gemm_bf16<0><<<g512, b256, 0, stream>>>(zb, WqP, bq, nullptr, bufA, nullptr, L_N, C_N, C_N);
  k_gemm_bf16<0><<<g512, b256, 0, stream>>>(zb, WkP, bk, nullptr, bufB, nullptr, L_N, C_N, C_N);
  k_gemm_bf16<0><<<g512, b256, 0, stream>>>(zb, WvP, bv, nullptr, bufC, nullptr, L_N, C_N, C_N);

  // 4) edge attention (3 passes with f32 atomics)
  int gEH = (E_N * H_N + 255) / 256;
  k_edge_score<<<gEH, b256, 0, stream>>>(bufA, bufB, row_idx, col_idx, pos_bias, score, mrow);
  k_zero_f32<<<(L_N * C_N + 255) / 256, b256, 0, stream>>>(bufA, L_N * C_N);  // attout=0 (q dead)
  k_edge_pexp<<<gEH, b256, 0, stream>>>(row_idx, mrow, score, den);
  int gEC4 = (int)(((size_t)E_N * (C_N / 4) + 255) / 256);
  k_edge_accum<<<gEC4, b256, 0, stream>>>(bufC, row_idx, col_idx, tcol_idx, dist, pos,
                                          col_pos, Wvec, bvec, score, den, bufA);

  // 5) att projection + residual: x1 = x + attout @ Wo + bo   (k dead -> bufB)
  k_f32_to_bf16<<<(L_N * C_N + 255) / 256, b256, 0, stream>>>(bufA, zb, L_N * C_N);
  k_gemm_bf16<1><<<g512, b256, 0, stream>>>(zb, WoP, bo, x, bufB, nullptr, L_N, C_N, C_N);

  // 6) FFN: z2 = LN2(x1); h = gelu(z2 @ W1 + b1) (bf16); out = x1 + h @ W2 + b2
  k_ln_bf16<<<L_N, b256, 0, stream>>>(bufB, ln2_g, ln2_b, zb);
  k_gemm_bf16<2><<<g1024, b256, 0, stream>>>(zb, W1P, b1, nullptr, nullptr, hb, L_N, HID_N, C_N);
  k_gemm_bf16<1><<<g512, b256, 0, stream>>>(hb, W2P, b2, bufB, out, nullptr, L_N, C_N, HID_N);
}
